// MultiHeadSelfAttention_89069031784540
// MI455X (gfx1250) — compile-verified, matrix operand order checked by reading, not run on hardware
//
#include <hip/hip_runtime.h>
#include <hip/hip_bf16.h>

// MI455X / gfx1250, wave32. All GEMMs via v_wmma_f32_16x16x32_bf16.

typedef __attribute__((ext_vector_type(16))) __bf16 v16bf;
typedef __attribute__((ext_vector_type(8)))  __bf16 bf16x8;
typedef __attribute__((ext_vector_type(8)))  float  v8f;
typedef __attribute__((ext_vector_type(4)))  int    v4i_;

#define B_  4
#define S_  2048
#define D_  1024
#define H_  16
#define HD_ 64

// ---- CDNA5 async global->LDS path (ASYNCcnt), with sync fallback ----------
#if __has_builtin(__builtin_amdgcn_global_load_async_to_lds_b128) && \
    __has_builtin(__builtin_amdgcn_s_wait_asynccnt)
#define ASYNC_LDS 1
__device__ __forceinline__ void async_cp16(const __bf16* g, __bf16* l) {
  __builtin_amdgcn_global_load_async_to_lds_b128(
      (__attribute__((address_space(1))) v4i_*)(void*)g,
      (__attribute__((address_space(3))) v4i_*)(void*)l, 0, 0);
}
#else
#define ASYNC_LDS 0
#endif

__device__ __forceinline__ v16bf make_frag(const __bf16* lo_p, const __bf16* hi_p) {
  bf16x8 lo = *(const bf16x8*)lo_p;   // K = 8*half + 0..7
  bf16x8 hi = *(const bf16x8*)hi_p;   // K = 16 + 8*half + 0..7
  return __builtin_shufflevector(lo, hi, 0,1,2,3,4,5,6,7,8,9,10,11,12,13,14,15);
}

// ---------------------------------------------------------------- f32 -> bf16
__global__ void cvt_f32_bf16(const float* __restrict__ in, __bf16* __restrict__ out, int n) {
  int i = blockIdx.x * blockDim.x + threadIdx.x;
  if (i < n) out[i] = (__bf16)in[i];
}

// ------------------------------------------------- C[M,N] = A[M,K] * W[N,K]^T
// Block tile 128x128, BK=64, 8 waves (4 down M x 2 across N); wave tile 32x64.
// 16 WMMAs per K-iteration.
__launch_bounds__(256)
__global__ void gemm_bf16_nt(const __bf16* __restrict__ A, const __bf16* __restrict__ W,
                             const float* __restrict__ bias,
                             __bf16* __restrict__ Cb, float* __restrict__ Cf,
                             int M, int N, int K, int write_f32)
{
  __shared__ __bf16 As[128][72];   // +8 pad: 144B row stride, 16B aligned
  __shared__ __bf16 Bs[128][72];
  const int tid  = threadIdx.x;
  const int wave = tid >> 5, lane = tid & 31;
  const int half = lane >> 4, l16 = lane & 15;
  const int bm = blockIdx.y * 128, bn = blockIdx.x * 128;
  const int wm = (wave & 3) * 32;        // 4 waves down M
  const int wn = (wave >> 2) * 64;       // 2 waves across N

  v8f acc[2][4] = {};

  const int sr = tid >> 1, sc = (tid & 1) * 32;   // staging: 2 threads/row, 64B each

  for (int kk = 0; kk < K; kk += 64) {
    const __bf16* srcA = A + (size_t)(bm + sr) * K + kk + sc;
    const __bf16* srcB = W + (size_t)(bn + sr) * K + kk + sc;
#if ASYNC_LDS
#pragma unroll
    for (int u = 0; u < 4; ++u) {
      async_cp16(srcA + 8 * u, &As[sr][sc + 8 * u]);
      async_cp16(srcB + 8 * u, &Bs[sr][sc + 8 * u]);
    }
    __builtin_amdgcn_s_wait_asynccnt(0);
#else
#pragma unroll
    for (int u = 0; u < 4; ++u) {
      *(bf16x8*)&As[sr][sc + 8 * u] = *(const bf16x8*)(srcA + 8 * u);
      *(bf16x8*)&Bs[sr][sc + 8 * u] = *(const bf16x8*)(srcB + 8 * u);
    }
#endif
    __syncthreads();

#pragma unroll
    for (int ks = 0; ks < 2; ++ks) {               // two K=32 steps per tile
      v16bf af[2], wf[4];
#pragma unroll
      for (int i = 0; i < 2; ++i) {
        int row = wm + i * 16 + l16;
        af[i] = make_frag(&As[row][32 * ks + 8 * half],
                          &As[row][32 * ks + 16 + 8 * half]);
      }
#pragma unroll
      for (int j = 0; j < 4; ++j) {
        int row = wn + j * 16 + l16;
        wf[j] = make_frag(&Bs[row][32 * ks + 8 * half],
                          &Bs[row][32 * ks + 16 + 8 * half]);
      }
#pragma unroll
      for (int i = 0; i < 2; ++i)
#pragma unroll
        for (int j = 0; j < 4; ++j)
          acc[i][j] = __builtin_amdgcn_wmma_f32_16x16x32_bf16(
              false, af[i], false, wf[j], (short)0, acc[i][j], false, false);
    }
    __syncthreads();
  }

#pragma unroll
  for (int i = 0; i < 2; ++i)
#pragma unroll
    for (int j = 0; j < 4; ++j) {
      int n = bn + wn + j * 16 + l16;
      float bv = bias ? bias[n] : 0.f;
#pragma unroll
      for (int rr = 0; rr < 8; ++rr) {
        int m = bm + wm + i * 16 + rr + 8 * half;  // C layout: M=rr+8*half, N=lane&15
        float v = acc[i][j][rr] + bv;
        if (write_f32) Cf[(size_t)m * N + n] = v;
        else           Cb[(size_t)m * N + n] = (__bf16)v;
      }
    }
}

// ---------------------------- RoPE on Q,K + repack; V stored transposed [HD,S]
__global__ void rope_pack(const __bf16* __restrict__ qb, const __bf16* __restrict__ kb2,
                          const __bf16* __restrict__ vb,
                          const float* __restrict__ cosT, const float* __restrict__ sinT,
                          __bf16* __restrict__ Qo, __bf16* __restrict__ Ko,
                          __bf16* __restrict__ VTo)
{
  size_t idx = (size_t)blockIdx.x * blockDim.x + threadIdx.x;  // over B*S*H*HD
  int hd = (int)(idx & (HD_ - 1));
  size_t row = idx >> 6;            // (b*S+s)*H + h
  int h = (int)(row & (H_ - 1));
  size_t bs = row >> 4;             // b*S + s
  int s = (int)(bs & (S_ - 1));
  int b = (int)(bs >> 11);

  float c  = cosT[s * HD_ + hd];
  float sn = sinT[s * HD_ + hd];
  int   hd2 = (hd < HD_ / 2) ? hd + HD_ / 2 : hd - HD_ / 2;
  float sgn = (hd < HD_ / 2) ? -1.f : 1.f;

  size_t base = bs * D_ + (size_t)h * HD_;
  float q  = (float)qb [base + hd],  qr = sgn * (float)qb [base + hd2];
  float k  = (float)kb2[base + hd],  kr = sgn * (float)kb2[base + hd2];

  size_t dst = (((size_t)b * H_ + h) * S_ + s) * HD_ + hd;
  Qo[dst] = (__bf16)(q * c + qr * sn);
  Ko[dst] = (__bf16)(k * c + kr * sn);
  VTo[(((size_t)b * H_ + h) * HD_ + hd) * S_ + s] = vb[base + hd];
}

// ------------------------------------------------ causal flash attention
// One wave per 16-row Q tile; 32-key blocks; online softmax; PV via WMMA.
__launch_bounds__(256)
__global__ void flash_attn(const __bf16* __restrict__ Q, const __bf16* __restrict__ Kk,
                           const __bf16* __restrict__ VT, __bf16* __restrict__ attn)
{
  __shared__ __bf16 Plds[8][16][40];   // per-wave P transpose staging
  const int wave = threadIdx.x >> 5, lane = threadIdx.x & 31;
  const int half = lane >> 4, l16 = lane & 15;
  const int bh = blockIdx.y;
  const int b = bh >> 4, h = bh & 15;
  const int qbase = blockIdx.x * 128 + wave * 16;

  const __bf16* Qp = Q  + (size_t)bh * S_ * HD_;
  const __bf16* Kp = Kk + (size_t)bh * S_ * HD_;
  const __bf16* Vp = VT + (size_t)bh * HD_ * S_;

  v16bf qf[2];
  {
    const __bf16* qrow = Qp + (size_t)(qbase + l16) * HD_;
#pragma unroll
    for (int f = 0; f < 2; ++f)
      qf[f] = make_frag(qrow + 32 * f + 8 * half, qrow + 32 * f + 16 + 8 * half);
  }

  float m_i[8], l_i[8];
#pragma unroll
  for (int r = 0; r < 8; ++r) { m_i[r] = -3.0e38f; l_i[r] = 0.f; }
  v8f o[4] = {};
  const float scale = 0.125f;  // HD^-0.5

  for (int kb = 0; kb <= qbase + 15; kb += 32) {
    __builtin_prefetch(Kp + (size_t)(kb + 32 + l16) * HD_, 0, 1);

    v8f sacc[2] = {};
#pragma unroll
    for (int j = 0; j < 2; ++j) {
      const __bf16* krow = Kp + (size_t)(kb + 16 * j + l16) * HD_;
      v16bf kf0 = make_frag(krow +      8 * half, krow + 16 + 8 * half);
      v16bf kf1 = make_frag(krow + 32 + 8 * half, krow + 48 + 8 * half);
      sacc[j] = __builtin_amdgcn_wmma_f32_16x16x32_bf16(
          false, qf[0], false, kf0, (short)0, sacc[j], false, false);
      sacc[j] = __builtin_amdgcn_wmma_f32_16x16x32_bf16(
          false, qf[1], false, kf1, (short)0, sacc[j], false, false);
    }

    float p0[8], p1[8], al[8];
#pragma unroll
    for (int r = 0; r < 8; ++r) {
      int q_abs = qbase + r + 8 * half;
      float s0 = sacc[0][r] * scale;
      float s1 = sacc[1][r] * scale;
      if (kb +      l16 > q_abs) s0 = -3.0e38f;   // causal mask
      if (kb + 16 + l16 > q_abs) s1 = -3.0e38f;
      float t = fmaxf(s0, s1);
#pragma unroll
      for (int msk = 1; msk < 16; msk <<= 1)      // row-max across N lanes
        t = fmaxf(t, __shfl_xor(t, msk, 32));
      float mn = fmaxf(m_i[r], t);
      al[r] = __expf(m_i[r] - mn);
      p0[r] = __expf(s0 - mn);
      p1[r] = __expf(s1 - mn);
      float rs = p0[r] + p1[r];
#pragma unroll
      for (int msk = 1; msk < 16; msk <<= 1)      // row-sum across N lanes
        rs += __shfl_xor(rs, msk, 32);
      l_i[r] = l_i[r] * al[r] + rs;
      m_i[r] = mn;
    }
#pragma unroll
    for (int c = 0; c < 4; ++c)
#pragma unroll
      for (int r = 0; r < 8; ++r)
        o[c][r] *= al[r];

    // transpose P (C-layout) -> A-fragment layout via per-wave LDS
#pragma unroll
    for (int r = 0; r < 8; ++r) {
      Plds[wave][r + 8 * half][l16]      = (__bf16)p0[r];
      Plds[wave][r + 8 * half][16 + l16] = (__bf16)p1[r];
    }
    __builtin_amdgcn_wave_barrier();
    v16bf pf = make_frag(&Plds[wave][l16][8 * half], &Plds[wave][l16][16 + 8 * half]);
    __builtin_amdgcn_wave_barrier();

#pragma unroll
    for (int c = 0; c < 4; ++c) {                 // O += P * V  (V read via VT rows)
      const __bf16* vrow = Vp + (size_t)(c * 16 + l16) * S_ + kb;
      v16bf vf = make_frag(vrow + 8 * half, vrow + 16 + 8 * half);
      o[c] = __builtin_amdgcn_wmma_f32_16x16x32_bf16(
          false, pf, false, vf, (short)0, o[c], false, false);
    }
  }

#pragma unroll
  for (int c = 0; c < 4; ++c)
#pragma unroll
    for (int r = 0; r < 8; ++r) {
      int s  = qbase + r + 8 * half;
      int hd = c * 16 + l16;
      attn[((size_t)b * S_ + s) * D_ + (size_t)h * HD_ + hd] = (__bf16)(o[c][r] / l_i[r]);
    }
}

// ---------------------------------------------------------------- launcher
extern "C" void kernel_launch(void* const* d_in, const int* in_sizes, int n_in,
                              void* d_out, int out_size, void* d_ws, size_t ws_size,
                              hipStream_t stream) {
  (void)in_sizes; (void)n_in; (void)out_size; (void)ws_size;

  const float* x    = (const float*)d_in[0];
  const float* cosT = (const float*)d_in[1];
  const float* sinT = (const float*)d_in[2];
  // d_in[3] = attn_mask (exactly causal; applied analytically in flash_attn)
  const float* Wq = (const float*)d_in[4];
  const float* bq = (const float*)d_in[5];
  const float* Wk = (const float*)d_in[6];
  const float* bk = (const float*)d_in[7];
  const float* Wv = (const float*)d_in[8];
  const float* bv = (const float*)d_in[9];
  const float* Wo = (const float*)d_in[10];
  const float* bo = (const float*)d_in[11];
  float* out = (float*)d_out;

  char* ws = (char*)d_ws;
  __bf16* xb   = (__bf16*)(ws);                    // 16 MB  [B*S, D]
  __bf16* wqb  = (__bf16*)(ws + (16u  << 20));     //  2 MB
  __bf16* wkb  = (__bf16*)(ws + (18u  << 20));
  __bf16* wvb  = (__bf16*)(ws + (20u  << 20));
  __bf16* wob  = (__bf16*)(ws + (22u  << 20));
  __bf16* qbuf = (__bf16*)(ws + (24u  << 20));     // 16 MB each
  __bf16* kbuf = (__bf16*)(ws + (40u  << 20));
  __bf16* vbuf = (__bf16*)(ws + (56u  << 20));
  __bf16* Qr   = (__bf16*)(ws + (72u  << 20));     // [B,H,S,HD]
  __bf16* Kr   = (__bf16*)(ws + (88u  << 20));     // [B,H,S,HD]
  __bf16* VT   = (__bf16*)(ws + (104u << 20));     // [B,H,HD,S]
  __bf16* attn = qbuf;                              // reuse (qbuf dead after rope)

  const int NX = B_ * S_ * D_;       // 8,388,608
  const int NW = D_ * D_;            // 1,048,576
  cvt_f32_bf16<<<NX / 256, 256, 0, stream>>>(x,  xb,  NX);
  cvt_f32_bf16<<<NW / 256, 256, 0, stream>>>(Wq, wqb, NW);
  cvt_f32_bf16<<<NW / 256, 256, 0, stream>>>(Wk, wkb, NW);
  cvt_f32_bf16<<<NW / 256, 256, 0, stream>>>(Wv, wvb, NW);
  cvt_f32_bf16<<<NW / 256, 256, 0, stream>>>(Wo, wob, NW);

  dim3 gg(D_ / 128, (B_ * S_) / 128);  // (8, 64)
  gemm_bf16_nt<<<gg, 256, 0, stream>>>(xb, wqb, bq, qbuf, nullptr, B_ * S_, D_, D_, 0);
  gemm_bf16_nt<<<gg, 256, 0, stream>>>(xb, wkb, bk, kbuf, nullptr, B_ * S_, D_, D_, 0);
  gemm_bf16_nt<<<gg, 256, 0, stream>>>(xb, wvb, bv, vbuf, nullptr, B_ * S_, D_, D_, 0);

  rope_pack<<<NX / 256, 256, 0, stream>>>(qbuf, kbuf, vbuf, cosT, sinT, Qr, Kr, VT);

  flash_attn<<<dim3(S_ / 128, B_ * H_), 256, 0, stream>>>(Qr, Kr, VT, attn);

  gemm_bf16_nt<<<gg, 256, 0, stream>>>(attn, wob, bo, nullptr, out, B_ * S_, D_, D_, 1);
}
